// DWAMFormer_45432164057332
// MI455X (gfx1250) — matrix operand order repaired and checked by reading, not compiled, not run on hardware
//
#include <hip/hip_runtime.h>
#include <hip/hip_bf16.h>

// ---------------------------------------------------------------------------
// Problem constants (DWAMFormer frame-merge): B=16, T=4000, C=512, MS=5
// ---------------------------------------------------------------------------
constexpr int BQ   = 16;
constexpr int TQ   = 4000;
constexpr int CQ   = 512;
constexpr int MSQ  = 5;
constexpr int TWQ  = TQ / MSQ;          // 800
constexpr int MROWS = BQ * TWQ;         // 12800  (GEMM M)
constexpr int DQ   = CQ * MSQ;          // 2560
constexpr int DRQ  = DQ * 2;            // 5120

typedef __attribute__((ext_vector_type(16))) __bf16 v16bf;
typedef __attribute__((ext_vector_type(8)))  __bf16 v8bf;
typedef __attribute__((ext_vector_type(8)))  float  v8f;

// ---------------------------------------------------------------------------
// fp32 -> bf16 conversion with transpose: src is K x N row-major,
// dst is N x K row-major (i.e. dst[n*K + k] = src[k*N + n]).
// Used once per launch on the weight matrices so the GEMM's B operand can be
// staged into LDS with contiguous 16-byte async copies (no LDS scatter).
// ---------------------------------------------------------------------------
__global__ void f32_to_bf16_transpose_kernel(const float* __restrict__ src,
                                             __bf16* __restrict__ dst,
                                             int K, int N) {
    int idx = blockIdx.x * blockDim.x + threadIdx.x;
    if (idx >= K * N) return;
    int n = idx / K, k = idx % K;
    dst[idx] = (__bf16)src[(size_t)k * N + n];
}

// ---------------------------------------------------------------------------
// Build flat activation matrix: flat[row, c*5+m] = x[b, w*5+m, c], bf16
// ---------------------------------------------------------------------------
__global__ void pack_flat_kernel(const float* __restrict__ x,
                                 __bf16* __restrict__ flat) {
    int idx = blockIdx.x * blockDim.x + threadIdx.x;   // over MROWS*CQ
    if (idx >= MROWS * CQ) return;
    int row = idx / CQ, c = idx % CQ;
    int b = row / TWQ, w = row % TWQ;
    const float* xp = x + ((size_t)b * TQ + (size_t)w * MSQ) * CQ + c;
    __bf16* fp = flat + (size_t)row * DQ + (size_t)c * MSQ;
#pragma unroll
    for (int m = 0; m < MSQ; ++m) fp[m] = (__bf16)xp[(size_t)m * CQ];
}

// ---------------------------------------------------------------------------
// Async global->LDS staging of a 128x32 bf16 tile (rows contiguous in k).
// Each of the 256 threads issues two GLOBAL_LOAD_ASYNC_TO_LDS_B128,
// tracked on ASYNCcnt (no VGPR round trip, no LDS scatter).
// ---------------------------------------------------------------------------
__device__ __forceinline__ void stage_tile_async(const __bf16* __restrict__ gbase,
                                                 int row_base, int ldk, int k0,
                                                 __bf16 (*lds)[32], int tid) {
#pragma unroll
    for (int cc = 0; cc < 2; ++cc) {
        int ch = tid * 2 + cc;              // 512 chunks of 8 bf16
        int r  = ch >> 2;
        int kc = (ch & 3) * 8;
        const void* g = gbase + (size_t)(row_base + r) * ldk + k0 + kc;
        unsigned l = (unsigned)(size_t)&lds[r][kc];   // LDS byte offset (addr[31:0])
        asm volatile("global_load_async_to_lds_b128 %0, %1, off"
                     :: "v"(l), "v"(g) : "memory");
    }
}

// ---------------------------------------------------------------------------
// Tiled bf16 WMMA GEMM: C = act(A @ Bt^T)
//   A : M x K row-major bf16
//   Bt: N x K row-major bf16 (pre-transposed weights)
// Block tile 128x128, BK=32, 256 threads = 8 waves (2x4), each wave computes
// a 64x32 tile via 4x2 v_wmma_f32_16x16x32_bf16. Double-buffered async
// global->LDS staging, one barrier per K-step.
// act: 0=none 1=relu 2=sigmoid.  Output: bf16 (Cb) or fp32+bias (Cf).
// ---------------------------------------------------------------------------
__global__ __launch_bounds__(256)
void gemm_wmma_bf16_kernel(const __bf16* __restrict__ A,
                           const __bf16* __restrict__ Bt,
                           __bf16* __restrict__ Cb,
                           float*  __restrict__ Cf,
                           const float* __restrict__ bias,
                           int M, int N, int K, int act) {
    __shared__ __bf16 lA[2][128][32];   // (m, k) tiles, double buffered
    __shared__ __bf16 lB[2][128][32];   // (n, k) tiles, double buffered

    const int tid  = threadIdx.x;
    const int wave = tid >> 5;
    const int lane = tid & 31;
    const int wm   = wave & 1;        // wave M index (0..1)
    const int wn   = wave >> 1;       // wave N index (0..3)
    const int half = lane >> 4;       // lane group (0..1)
    const int r    = lane & 15;

    const int rb = blockIdx.y * 128;  // block row base
    const int nb = blockIdx.x * 128;  // block col base

    v8f acc[4][2];
#pragma unroll
    for (int mt = 0; mt < 4; ++mt)
#pragma unroll
        for (int nt = 0; nt < 2; ++nt)
#pragma unroll
            for (int i = 0; i < 8; ++i) acc[mt][nt][i] = 0.0f;

    const int nk = K >> 5;  // K / 32 (all K are multiples of 32)

    // Prologue: stage tile 0 into buffer 0.
    stage_tile_async(A,  rb, K, 0, lA[0], tid);
    stage_tile_async(Bt, nb, K, 0, lB[0], tid);

    for (int kt = 0; kt < nk; ++kt) {
        const int buf = kt & 1;
        // Wait for our own async copies, then rendezvous with the block.
        asm volatile("s_wait_asynccnt 0x0" ::: "memory");
        __syncthreads();

        // Kick off next tile into the other buffer (overlaps with WMMA below).
        if (kt + 1 < nk) {
            int k0n = (kt + 1) << 5;
            stage_tile_async(A,  rb, K, k0n, lA[buf ^ 1], tid);
            stage_tile_async(Bt, nb, K, k0n, lB[buf ^ 1], tid);
            __builtin_prefetch(A + (size_t)(rb + (tid >> 1)) * K + k0n, 0, 1);
        }

        // ---- build fragments per ISA VGPR layout ----
        // A 16x32 bf16: lanes 0-15 hold K 0..7 then 16..23; lanes 16-31 hold
        // K 8..15 then 24..31 (row M = lane%16).
        v16bf afrag[4];
#pragma unroll
        for (int mt = 0; mt < 4; ++mt) {
            int arow = wm * 64 + mt * 16 + r;
            v8bf lo = *reinterpret_cast<const v8bf*>(&lA[buf][arow][half * 8]);
            v8bf hi = *reinterpret_cast<const v8bf*>(&lA[buf][arow][16 + half * 8]);
            afrag[mt] = __builtin_shufflevector(lo, hi,
                0, 1, 2, 3, 4, 5, 6, 7, 8, 9, 10, 11, 12, 13, 14, 15);
        }
        // B 32x16 bf16: lanes 0-15 hold K 0..7 then 16..23 of column N=lane;
        // lanes 16-31 hold K 8..15 then 24..31.  lB rows are contiguous in k.
        v16bf bfrag[2];
#pragma unroll
        for (int nt = 0; nt < 2; ++nt) {
            int bcol = wn * 32 + nt * 16 + r;
            v8bf lo = *reinterpret_cast<const v8bf*>(&lB[buf][bcol][half * 8]);
            v8bf hi = *reinterpret_cast<const v8bf*>(&lB[buf][bcol][16 + half * 8]);
            bfrag[nt] = __builtin_shufflevector(lo, hi,
                0, 1, 2, 3, 4, 5, 6, 7, 8, 9, 10, 11, 12, 13, 14, 15);
        }

#pragma unroll
        for (int mt = 0; mt < 4; ++mt)
#pragma unroll
            for (int nt = 0; nt < 2; ++nt)
                acc[mt][nt] = __builtin_amdgcn_wmma_f32_16x16x32_bf16(
                    false, afrag[mt], false, bfrag[nt],
                    (short)0, acc[mt][nt], false, false);
        // No trailing barrier needed: fragments for this step were already
        // consumed into registers before any wave can pass the next barrier
        // and overwrite this buffer (double buffering).
    }

    // ---- epilogue: element i of v8f -> row (half*8 + i), col r of the tile --
#pragma unroll
    for (int mt = 0; mt < 4; ++mt) {
#pragma unroll
        for (int nt = 0; nt < 2; ++nt) {
            int gr0 = rb + wm * 64 + mt * 16 + half * 8;
            int gc  = nb + wn * 32 + nt * 16 + r;
#pragma unroll
            for (int i = 0; i < 8; ++i) {
                float v = acc[mt][nt][i];
                if (act == 1)      v = fmaxf(v, 0.0f);
                else if (act == 2) v = 1.0f / (1.0f + __expf(-v));
                size_t off = (size_t)(gr0 + i) * N + gc;
                if (Cf) Cf[off] = v + (bias ? bias[gc] : 0.0f);
                else    Cb[off] = (__bf16)v;
            }
        }
    }
}

// ---------------------------------------------------------------------------
// Softmax over window (m=0..4) + weighted pool against original fp32 x.
// ---------------------------------------------------------------------------
__global__ void pool_softmax_kernel(const float* __restrict__ x,
                                    const __bf16* __restrict__ Y,
                                    __bf16* __restrict__ pooled) {
    int idx = blockIdx.x * blockDim.x + threadIdx.x;   // over MROWS*CQ
    if (idx >= MROWS * CQ) return;
    int row = idx / CQ, c = idx % CQ;
    int b = row / TWQ, w = row % TWQ;
    const __bf16* yp = Y + (size_t)row * DQ + (size_t)c * MSQ;
    const float*  xp = x + ((size_t)b * TQ + (size_t)w * MSQ) * CQ + c;
    float yv[MSQ];
    float mx = -3.0e38f;
#pragma unroll
    for (int m = 0; m < MSQ; ++m) { yv[m] = (float)yp[m]; mx = fmaxf(mx, yv[m]); }
    float s = 0.0f;
#pragma unroll
    for (int m = 0; m < MSQ; ++m) { yv[m] = __expf(yv[m] - mx); s += yv[m]; }
    float inv = 1.0f / s;
    float accv = 0.0f;
#pragma unroll
    for (int m = 0; m < MSQ; ++m) accv += yv[m] * inv * xp[(size_t)m * CQ];
    pooled[idx] = (__bf16)accv;
}

// ---------------------------------------------------------------------------
// LayerNorm over the last (512) dim: one block (256 thr) per row.
// ---------------------------------------------------------------------------
__global__ __launch_bounds__(256)
void layernorm_kernel(const float* __restrict__ h,
                      const float* __restrict__ g,
                      const float* __restrict__ bparm,
                      float* __restrict__ out) {
    __shared__ float ssum[256];
    __shared__ float ssq[256];
    int row = blockIdx.x;
    int t = threadIdx.x;
    const float* hp = h + (size_t)row * CQ;
    float v0 = hp[t], v1 = hp[t + 256];
    ssum[t] = v0 + v1;
    ssq[t]  = v0 * v0 + v1 * v1;
    __syncthreads();
    for (int o = 128; o > 0; o >>= 1) {
        if (t < o) { ssum[t] += ssum[t + o]; ssq[t] += ssq[t + o]; }
        __syncthreads();
    }
    float mean = ssum[0] * (1.0f / CQ);
    float var  = ssq[0] * (1.0f / CQ) - mean * mean;
    float rs   = rsqrtf(var + 1e-5f);
    float* op = out + (size_t)row * CQ;
    op[t]       = (v0 - mean) * rs * g[t]       + bparm[t];
    op[t + 256] = (v1 - mean) * rs * g[t + 256] + bparm[t + 256];
}

// ---------------------------------------------------------------------------
// Launch
// ---------------------------------------------------------------------------
extern "C" void kernel_launch(void* const* d_in, const int* in_sizes, int n_in,
                              void* d_out, int out_size, void* d_ws, size_t ws_size,
                              hipStream_t stream) {
    (void)in_sizes; (void)n_in; (void)out_size; (void)ws_size;
    const float* x    = (const float*)d_in[0];   // (16, 4000, 512)
    const float* w1   = (const float*)d_in[1];   // (2560, 5120)
    const float* w2   = (const float*)d_in[2];   // (5120, 2560)
    const float* fcw  = (const float*)d_in[3];   // (512, 512)
    const float* fcb  = (const float*)d_in[4];   // (512,)
    const float* lng  = (const float*)d_in[5];   // (512,)
    const float* lnb  = (const float*)d_in[6];   // (512,)
    float* out = (float*)d_out;                  // (16, 800, 512)

    // Workspace layout (all chunks multiple of 256 B)
    char* ws = (char*)d_ws;
    size_t off = 0;
    __bf16* flatB = (__bf16*)(ws + off); off += (size_t)MROWS * DQ * 2;   // 65.5 MB
    __bf16* w1t   = (__bf16*)(ws + off); off += (size_t)DQ * DRQ * 2;     // 26.2 MB (5120 x 2560)
    __bf16* w2t   = (__bf16*)(ws + off); off += (size_t)DRQ * DQ * 2;     // 26.2 MB (2560 x 5120)
    __bf16* fcwt  = (__bf16*)(ws + off); off += (size_t)CQ * CQ * 2;      // 0.5 MB  (512 x 512)
    __bf16* Hb    = (__bf16*)(ws + off); off += (size_t)MROWS * DRQ * 2;  // 131 MB
    __bf16* Yb    = (__bf16*)(ws + off); off += (size_t)MROWS * DQ * 2;   // 65.5 MB
    __bf16* poolB = (__bf16*)(ws + off); off += (size_t)MROWS * CQ * 2;   // 13.1 MB
    float*  fcout = (float*) (ws + off); off += (size_t)MROWS * CQ * 4;   // 26.2 MB

    // 1) weight conversions to bf16, pre-transposed to (N x K)
    {
        int n1 = DQ * DRQ, n2 = DRQ * DQ, n3 = CQ * CQ;
        f32_to_bf16_transpose_kernel<<<(n1 + 255) / 256, 256, 0, stream>>>(w1, w1t, DQ, DRQ);
        f32_to_bf16_transpose_kernel<<<(n2 + 255) / 256, 256, 0, stream>>>(w2, w2t, DRQ, DQ);
        f32_to_bf16_transpose_kernel<<<(n3 + 255) / 256, 256, 0, stream>>>(fcw, fcwt, CQ, CQ);
    }
    // 2) pack flat activations
    {
        int n = MROWS * CQ;
        pack_flat_kernel<<<(n + 255) / 256, 256, 0, stream>>>(x, flatB);
    }
    // 3) H = relu(flat @ w1)   (12800 x 5120)
    {
        dim3 grid(DRQ / 128, MROWS / 128);
        gemm_wmma_bf16_kernel<<<grid, 256, 0, stream>>>(
            flatB, w1t, Hb, nullptr, nullptr, MROWS, DRQ, DQ, /*relu*/1);
    }
    // 4) Y = sigmoid(H @ w2)   (12800 x 2560)
    {
        dim3 grid(DQ / 128, MROWS / 128);
        gemm_wmma_bf16_kernel<<<grid, 256, 0, stream>>>(
            Hb, w2t, Yb, nullptr, nullptr, MROWS, DQ, DRQ, /*sigmoid*/2);
    }
    // 5) softmax-over-window pooling
    {
        int n = MROWS * CQ;
        pool_softmax_kernel<<<(n + 255) / 256, 256, 0, stream>>>(x, Yb, poolB);
    }
    // 6) fcout = pooled @ fc_w + fc_b   (12800 x 512, fp32)
    {
        dim3 grid(CQ / 128, MROWS / 128);
        gemm_wmma_bf16_kernel<<<grid, 256, 0, stream>>>(
            poolB, fcwt, nullptr, fcout, fcb, MROWS, CQ, CQ, /*none*/0);
    }
    // 7) LayerNorm -> output
    layernorm_kernel<<<MROWS, 256, 0, stream>>>(fcout, lng, lnb, out);
}
